// MultiHeadSelfAttention_84086869721667
// MI455X (gfx1250) — compile-verified, matrix operand order checked by reading, not run on hardware
//
#include <hip/hip_runtime.h>

// ---- CDNA5 WMMA vector types ----
typedef __attribute__((ext_vector_type(16))) __bf16 v16bf;
typedef __attribute__((ext_vector_type(8)))  __bf16 v8bf;
typedef __attribute__((ext_vector_type(8)))  float  v8f;

#define NUM_HEADS 16
#define ATTN_DIM  1024
#define VALUES_DIM 1024
#define INPUT_DIM 1024
#define SEQ_T     2048
#define BATCH     4
#define C3        (2 * ATTN_DIM + VALUES_DIM)   // 3072
#define C2        (2 * ATTN_DIM)                // 2048 (Q|K in proj buffer)
#define DH        (ATTN_DIM / NUM_HEADS)        // 64
#define MROWS     (BATCH * SEQ_T)               // 8192

static __device__ __forceinline__ v8f wmma_bf16(const v16bf& a, const v16bf& b, const v8f& c) {
    return __builtin_amdgcn_wmma_f32_16x16x32_bf16(false, a, false, b, (short)0, c, false, false);
}

// 16x32 fragment (bf16): elements 0..7 = p[0..7], 8..15 = p[16..23]; both 16B aligned.
static __device__ __forceinline__ v16bf load_frag(const __bf16* p) {
    v8bf lo = *(const v8bf*)p;
    v8bf hi = *(const v8bf*)(p + 16);
    return __builtin_shufflevector(lo, hi, 0,1,2,3,4,5,6,7,8,9,10,11,12,13,14,15);
}

// ============================================================================
// Pre-pass 1: f32 -> bf16 elementwise (inputs)
// ============================================================================
__global__ __launch_bounds__(256)
void cvt_bf16_kernel(const float* __restrict__ X, __bf16* __restrict__ Y, int n4) {
    int i = (blockIdx.x * 256 + threadIdx.x);
    if (i < n4) {
        const float4 v = ((const float4*)X)[i];
        __bf16* y = Y + (size_t)i * 4;
        y[0] = (__bf16)v.x; y[1] = (__bf16)v.y; y[2] = (__bf16)v.z; y[3] = (__bf16)v.w;
    }
}

// ============================================================================
// Pre-pass 2: W[K][N] f32 -> Wt[N][K] bf16 (coalesced read, scattered write)
// ============================================================================
__global__ __launch_bounds__(256)
void transpose_bf16_kernel(const float* __restrict__ W, __bf16* __restrict__ Wt,
                           int K, int N) {
    size_t idx = (size_t)blockIdx.x * 256 + threadIdx.x;
    if (idx < (size_t)K * N) {
        int k = (int)(idx / N), n = (int)(idx % N);
        Wt[(size_t)n * K + k] = (__bf16)W[idx];
    }
}

// ============================================================================
// GEMM core: 1 wave -> 32 rows x 64 cols, all-bf16, K-step 32, 8 wmma/step.
// A[M][K] bf16, Wt[N][K] bf16 (pre-transposed). All loads = b128.
// ============================================================================
__device__ __forceinline__ void gemm_core(const __bf16* __restrict__ A,
                                          const __bf16* __restrict__ Wt,
                                          int K, int rowBase, int colBase,
                                          int l16, int hl, v8f acc[2][4]) {
#pragma unroll
    for (int s = 0; s < 2; ++s)
#pragma unroll
        for (int j = 0; j < 4; ++j) acc[s][j] = v8f{};

    const __bf16* a0 = A + (size_t)(rowBase + l16) * K + hl * 8;
    const __bf16* a1 = A + (size_t)(rowBase + 16 + l16) * K + hl * 8;
    const __bf16* wp[4];
#pragma unroll
    for (int j = 0; j < 4; ++j)
        wp[j] = Wt + (size_t)(colBase + j * 16 + l16) * K + hl * 8;

    for (int k0 = 0; k0 < K; k0 += 32) {
        __builtin_prefetch(a0 + k0 + 128, 0, 0);   // global_prefetch_b8
        v16bf af0 = load_frag(a0 + k0);
        v16bf af1 = load_frag(a1 + k0);
#pragma unroll
        for (int j = 0; j < 4; ++j) {
            v16bf bf = load_frag(wp[j] + k0);
            acc[0][j] = wmma_bf16(af0, bf, acc[0][j]);
            acc[1][j] = wmma_bf16(af1, bf, acc[1][j]);
        }
    }
}

// ============================================================================
// GEMM 1: x_bf16[8192,1024] * Wqkv_t -> Q|K (pre-scaled q) into proj[8192,2048],
// V written transposed into vt[B,H,64,T].
// ============================================================================
__global__ __launch_bounds__(128)
void gemm_qkv_kernel(const __bf16* __restrict__ A, const __bf16* __restrict__ Wt,
                     const float* __restrict__ bias,
                     __bf16* __restrict__ proj, __bf16* __restrict__ vt) {
    const int tid = threadIdx.x, wave = tid >> 5, lane = tid & 31;
    const int l16 = lane & 15, hl = lane >> 4;
    const int rowBase = blockIdx.x * 128 + wave * 32;
    const int colBase = blockIdx.y * 64;

    v8f acc[2][4];
    gemm_core(A, Wt, INPUT_DIM, rowBase, colBase, l16, hl, acc);

#pragma unroll
    for (int s = 0; s < 2; ++s)
#pragma unroll
        for (int j = 0; j < 4; ++j) {
            int col = colBase + j * 16 + l16;
            float bj = bias[col];
            float sc = (col < ATTN_DIM) ? 0.125f : 1.0f;   // q / sqrt(dh)
#pragma unroll
            for (int v = 0; v < 8; ++v) {
                int rr = rowBase + s * 16 + v + 8 * hl;
                float val = (acc[s][j][v] + bj) * sc;
                if (col < C2) {
                    proj[(size_t)rr * C2 + col] = (__bf16)val;
                } else {
                    int c2 = col - C2, h = c2 >> 6, d = c2 & 63;
                    int b = rr >> 11, t = rr & (SEQ_T - 1);
                    vt[(((size_t)b * NUM_HEADS + h) * DH + d) * SEQ_T + t] = (__bf16)val;
                }
            }
        }
}

// ============================================================================
// GEMM 3: attn_bf16[8192,1024] * Wout_t -> out f32 [8192,1024]
// ============================================================================
__global__ __launch_bounds__(128)
void gemm_out_kernel(const __bf16* __restrict__ A, const __bf16* __restrict__ Wt,
                     const float* __restrict__ bias, float* __restrict__ out) {
    const int tid = threadIdx.x, wave = tid >> 5, lane = tid & 31;
    const int l16 = lane & 15, hl = lane >> 4;
    const int rowBase = blockIdx.x * 128 + wave * 32;
    const int colBase = blockIdx.y * 64;

    v8f acc[2][4];
    gemm_core(A, Wt, VALUES_DIM, rowBase, colBase, l16, hl, acc);

#pragma unroll
    for (int s = 0; s < 2; ++s)
#pragma unroll
        for (int j = 0; j < 4; ++j) {
            int col = colBase + j * 16 + l16;
            float bj = bias[col];
#pragma unroll
            for (int v = 0; v < 8; ++v) {
                int rr = rowBase + s * 16 + v + 8 * hl;
                out[(size_t)rr * INPUT_DIM + col] = acc[s][j][v] + bj;
            }
        }
}

// ============================================================================
// Flash attention. proj[B,T,2048] = Q|K bf16 (q pre-scaled), vt[B,H,64,T] bf16.
// 4 waves x 16 q-rows; keys streamed in blocks of 32 with online softmax.
// Score tile transposed through LDS (needed for P A-layout anyway), so the
// row reductions are in-lane + one shfl_xor(16); m/ell are scalars per lane.
// ============================================================================
__global__ __launch_bounds__(128)
void flash_attn_kernel(const __bf16* __restrict__ proj,
                       const __bf16* __restrict__ vt,
                       const long long* __restrict__ mask,
                       __bf16* __restrict__ attn_out)
{
    __shared__ float ptile[4][16][40];   // per-wave 16x32 f32 tile (stride 160B)

    const int tid = threadIdx.x, wave = tid >> 5, lane = tid & 31;
    const int l16 = lane & 15, hl = lane >> 4;
    const int b = blockIdx.z, h = blockIdx.y;
    const int qbase = blockIdx.x * 64 + wave * 16;

    const __bf16* qp = proj + (size_t)b * SEQ_T * C2 + h * DH;
    const __bf16* kp = qp + ATTN_DIM;
    const __bf16* vb = vt + ((size_t)b * NUM_HEADS + h) * DH * SEQ_T;
    const long long* mrow = mask + (size_t)b * SEQ_T;

    // Q fragments (two 16x32 slices of d=0..63)
    const __bf16* q0 = qp + (size_t)(qbase + l16) * C2 + hl * 8;
    v16bf aq0 = load_frag(q0);
    v16bf aq1 = load_frag(q0 + 32);

    v8f acc[4];
#pragma unroll
    for (int j = 0; j < 4; ++j) acc[j] = v8f{};
    float m_run = -1.0e30f, ell = 0.0f;    // stats for row l16 (row-layout)

    for (int kb = 0; kb < SEQ_T; kb += 32) {
        // ---- S = Q K^T, C-layout (two 16x16 tiles), mask applied per key col
        v8f s[2];
#pragma unroll
        for (int j = 0; j < 2; ++j) {
            int key = kb + j * 16 + l16;
            const __bf16* kr = kp + (size_t)key * C2 + hl * 8;
            v8f c = v8f{};
            c = wmma_bf16(aq0, load_frag(kr), c);
            c = wmma_bf16(aq1, load_frag(kr + 32), c);
            if (mrow[key] <= 0) {
#pragma unroll
                for (int v = 0; v < 8; ++v) c[v] = -1.0e30f;
            }
            s[j] = c;
        }

        // ---- transpose S through LDS (C-layout -> row-per-lane / A-layout) --
        __syncthreads();
#pragma unroll
        for (int j = 0; j < 2; ++j)
#pragma unroll
            for (int v = 0; v < 8; ++v)
                ptile[wave][v + 8 * hl][j * 16 + l16] = s[j][v];
        __syncthreads();

        float sf[16];
        {
            const float* rowp = ptile[wave][l16];
            const float4 f0 = *(const float4*)(rowp + 8 * hl);
            const float4 f1 = *(const float4*)(rowp + 8 * hl + 4);
            const float4 f2 = *(const float4*)(rowp + 16 + 8 * hl);
            const float4 f3 = *(const float4*)(rowp + 16 + 8 * hl + 4);
            sf[0]=f0.x; sf[1]=f0.y; sf[2]=f0.z; sf[3]=f0.w;
            sf[4]=f1.x; sf[5]=f1.y; sf[6]=f1.z; sf[7]=f1.w;
            sf[8]=f2.x; sf[9]=f2.y; sf[10]=f2.z; sf[11]=f2.w;
            sf[12]=f3.x; sf[13]=f3.y; sf[14]=f3.z; sf[15]=f3.w;
        }

        // ---- online softmax for row l16: in-lane reduce + 1 cross-half shfl
        float rm = sf[0];
#pragma unroll
        for (int i = 1; i < 16; ++i) rm = fmaxf(rm, sf[i]);
        rm = fmaxf(rm, __shfl_xor(rm, 16, 32));
        float nm = fmaxf(m_run, rm);
        float corr = __expf(m_run - nm);
        float rs = 0.0f;
        v16bf pa;
#pragma unroll
        for (int i = 0; i < 16; ++i) {
            float p = __expf(sf[i] - nm);
            rs += p;
            pa[i] = (__bf16)p;       // already A-fragment element order
        }
        rs += __shfl_xor(rs, 16, 32);
        ell = ell * corr + rs;
        m_run = nm;

        // ---- rescale accumulators: corr for row v+8*hl lives at lane v+8*hl
#pragma unroll
        for (int v = 0; v < 8; ++v) {
            float cv = __shfl(corr, v + 8 * hl, 32);
#pragma unroll
            for (int j = 0; j < 4; ++j) acc[j][v] *= cv;
        }

        // ---- O += P * V  (V^T layout -> contiguous b128 fragment loads) ----
#pragma unroll
        for (int j = 0; j < 4; ++j) {
            const __bf16* vr = vb + (size_t)(j * 16 + l16) * SEQ_T + kb + hl * 8;
            acc[j] = wmma_bf16(pa, load_frag(vr), acc[j]);
        }
    }

    // ---- normalize and store bf16 output [B,T,1024] ----
#pragma unroll
    for (int v = 0; v < 8; ++v) {
        float er = __shfl(ell, v + 8 * hl, 32);
        float inv = 1.0f / er;
        int t = qbase + v + 8 * hl;
#pragma unroll
        for (int j = 0; j < 4; ++j) {
            attn_out[((size_t)b * SEQ_T + t) * VALUES_DIM + h * DH + j * 16 + l16] =
                (__bf16)(acc[j][v] * inv);
        }
    }
}

// ============================================================================
// Launch: pre-passes -> GEMM(qkv) -> flash attention -> GEMM(out)
// ============================================================================
extern "C" void kernel_launch(void* const* d_in, const int* in_sizes, int n_in,
                              void* d_out, int out_size, void* d_ws, size_t ws_size,
                              hipStream_t stream) {
    const float*     inputs = (const float*)d_in[0];       // [4,2048,1024] f32
    const long long* mask   = (const long long*)d_in[1];   // [4,2048] i64
    const float*     W_qkv  = (const float*)d_in[2];       // [1024,3072] f32
    const float*     b_qkv  = (const float*)d_in[3];       // [3072] f32
    const float*     W_out  = (const float*)d_in[4];       // [1024,1024] f32
    const float*     b_out  = (const float*)d_in[5];       // [1024] f32
    float*           out    = (float*)d_out;               // [4,2048,1024] f32

    // workspace layout (all bf16)
    __bf16* x_bf16  = (__bf16*)d_ws;                               // 16 MB
    __bf16* wqkv_t  = x_bf16 + (size_t)MROWS * INPUT_DIM;          //  6 MB
    __bf16* wout_t  = wqkv_t + (size_t)C3 * INPUT_DIM;             //  2 MB
    __bf16* proj    = wout_t + (size_t)INPUT_DIM * VALUES_DIM;     // 32 MB (Q|K)
    __bf16* vt      = proj   + (size_t)MROWS * C2;                 // 16 MB (V^T)
    __bf16* attn    = vt     + (size_t)BATCH * NUM_HEADS * DH * SEQ_T; // 16 MB

    // pre-passes
    {
        int n4 = (MROWS * INPUT_DIM) / 4;
        cvt_bf16_kernel<<<(n4 + 255) / 256, 256, 0, stream>>>(inputs, x_bf16, n4);
        size_t nq = (size_t)INPUT_DIM * C3;
        transpose_bf16_kernel<<<(unsigned)((nq + 255) / 256), 256, 0, stream>>>(
            W_qkv, wqkv_t, INPUT_DIM, C3);
        size_t no = (size_t)VALUES_DIM * INPUT_DIM;
        transpose_bf16_kernel<<<(unsigned)((no + 255) / 256), 256, 0, stream>>>(
            W_out, wout_t, VALUES_DIM, INPUT_DIM);
    }
    // 1) QKV projection (q pre-scaled; V written transposed)
    {
        dim3 grid(MROWS / 128, C3 / 64);
        gemm_qkv_kernel<<<grid, 128, 0, stream>>>(x_bf16, wqkv_t, b_qkv, proj, vt);
    }
    // 2) Flash attention
    {
        dim3 grid(SEQ_T / 64, NUM_HEADS, BATCH);
        flash_attn_kernel<<<grid, 128, 0, stream>>>(proj, vt, mask, attn);
    }
    // 3) Output projection
    {
        dim3 grid(MROWS / 128, INPUT_DIM / 64);
        gemm_out_kernel<<<grid, 128, 0, stream>>>(attn, wout_t, b_out, out);
    }
}
